// ChordMixerBlock_36112085025532
// MI455X (gfx1250) — matrix operand order repaired and checked by reading, not compiled, not run on hardware
//
#include <hip/hip_runtime.h>
#include <math.h>

typedef __attribute__((ext_vector_type(4)))  __bf16 v4bf;
typedef __attribute__((ext_vector_type(8)))  __bf16 v8bf;
typedef __attribute__((ext_vector_type(16))) __bf16 v16bf;
typedef __attribute__((ext_vector_type(8)))  float  v8f;

union ABFrag { v16bf v; v8bf h[2]; };

#define L_LEN   16384
#define D_DIM   256
#define H_DIM   512
#define M_TILE  64

// ---------------------------------------------------------------------------
// Prep: convert W1 [256x512] and W2 [512x256] (row-major fp32, (k,n) at k*N+n)
// into bf16 B-fragments for v_wmma_f32_16x16x32_bf16.
// B fragment (32x16, KxN): lane holds col = lane&15, K-half = lane>>4;
// lane's 16 elements in order K = khalf*16 + 0..15. Fragment stored as
// 32 lanes x 16 bf16 contiguous (1 KB), fragment index = kc*NT + nt.
// ---------------------------------------------------------------------------
__global__ void prep_weights_kernel(const float* __restrict__ w1,
                                    const float* __restrict__ w2,
                                    __bf16* __restrict__ w1b,
                                    __bf16* __restrict__ w2b) {
  int wave = (blockIdx.x * blockDim.x + threadIdx.x) >> 5;
  int lane = threadIdx.x & 31;
  if (wave >= 512) return;

  const float* src;
  __bf16* dst;
  int kc, nt, N;
  if (wave < 256) {            // W1: 8 K-chunks x 32 N-tiles
    kc = wave >> 5; nt = wave & 31;
    src = w1; N = H_DIM;
    dst = w1b + (size_t)wave * 512;
  } else {                     // W2: 16 K-chunks x 16 N-tiles
    int f = wave - 256;
    kc = f >> 4; nt = f & 15;
    src = w2; N = D_DIM;
    dst = w2b + (size_t)f * 512;
  }
  int col = nt * 16 + (lane & 15);
  int k0  = kc * 32 + (lane >> 4) * 16;

  v8bf lo, hi;
  #pragma unroll
  for (int j = 0; j < 8; ++j) lo[j] = (__bf16)src[(size_t)(k0 + j) * N + col];
  #pragma unroll
  for (int j = 0; j < 8; ++j) hi[j] = (__bf16)src[(size_t)(k0 + 8 + j) * N + col];

  *(v8bf*)(dst + lane * 16)     = lo;
  *(v8bf*)(dst + lane * 16 + 8) = hi;
}

// ---------------------------------------------------------------------------
// Fused: x@W1+b1 -> gelu -> @W2+b2 -> rotate_chord scatter + residual.
// One block = 64 tokens (4 row-blocks), 8 waves. Each wave owns a column
// group and ALL 4 row-blocks, so every weight fragment is loaded exactly
// once per block and reused across 4 A-fragments (min L2 weight traffic).
// ---------------------------------------------------------------------------
__global__ void __launch_bounds__(256)
chordmixer_fused_kernel(const float* __restrict__ data,
                        const float* __restrict__ b1,
                        const float* __restrict__ b2,
                        const __bf16* __restrict__ w1b,
                        const __bf16* __restrict__ w2b,
                        float* __restrict__ out) {
  __shared__ __bf16 sA[M_TILE][D_DIM + 8];   // input tile, bf16   (~34 KB)
  __shared__ __bf16 sH[M_TILE][H_DIM + 8];   // hidden tile, bf16  (~67 KB)

  const int tid = threadIdx.x;
  const int t0  = blockIdx.x * M_TILE;       // global token base

  // ---- stage input tile: coalesced float4 loads, cvt to bf16 into LDS ----
  const float* dptr = data + (size_t)t0 * D_DIM;
  #pragma unroll
  for (int it = 0; it < 16; ++it) {
    int e = tid * 4 + it * 1024;             // flat element in 64x256 tile
    float4 v = *(const float4*)(dptr + e);
    int row = e >> 8;
    int col = e & 255;
    v4bf p;
    p[0] = (__bf16)v.x; p[1] = (__bf16)v.y;
    p[2] = (__bf16)v.z; p[3] = (__bf16)v.w;
    *(v4bf*)&sA[row][col] = p;
  }
  __syncthreads();

  const int wv   = tid >> 5;                 // wave id 0..7 == column group
  const int lane = tid & 31;
  const int rlan = lane & 15;                // A row / D col within tile
  const int kb   = (lane >> 4) * 16;         // A-fragment K sub-base

  // ------------------------- GEMM1: h = x @ W1 ---------------------------
  // wave wv owns nt = wv*4 + j (j=0..3), row-blocks rb=0..3.
  v8f acc1[4][4] = {};                       // [j][rb]
  #pragma unroll
  for (int kc = 0; kc < 8; ++kc) {
    ABFrag a[4];
    #pragma unroll
    for (int rb = 0; rb < 4; ++rb) {
      a[rb].h[0] = *(const v8bf*)&sA[rb * 16 + rlan][kc * 32 + kb];
      a[rb].h[1] = *(const v8bf*)&sA[rb * 16 + rlan][kc * 32 + kb + 8];
    }
    #pragma unroll
    for (int j = 0; j < 4; ++j) {
      int nt = wv * 4 + j;
      const __bf16* fp = w1b + ((size_t)(kc * 32 + nt)) * 512 + lane * 16;
      ABFrag b;
      b.h[0] = *(const v8bf*)fp;
      b.h[1] = *(const v8bf*)(fp + 8);
      #pragma unroll
      for (int rb = 0; rb < 4; ++rb)
        acc1[j][rb] = __builtin_amdgcn_wmma_f32_16x16x32_bf16(
            false, a[rb].v, false, b.v, (short)0, acc1[j][rb], false, false);
    }
  }

  // ---- bias + exact GELU, write hidden tile to LDS as bf16 ----
  #pragma unroll
  for (int j = 0; j < 4; ++j) {
    int c = (wv * 4 + j) * 16 + rlan;
    float bias = b1[c];
    #pragma unroll
    for (int rb = 0; rb < 4; ++rb) {
      int rbase = rb * 16 + (lane >> 4) * 8;
      #pragma unroll
      for (int r = 0; r < 8; ++r) {
        float x = acc1[j][rb][r] + bias;
        float g = 0.5f * x * (1.0f + erff(x * 0.70710678118654752f));
        sH[rbase + r][c] = (__bf16)g;
      }
    }
  }
  __syncthreads();

  // ------------------------- GEMM2: y = h @ W2 ---------------------------
  // wave wv owns nt = wv*2 + j (j=0..1), row-blocks rb=0..3.
  v8f acc2[2][4] = {};
  #pragma unroll
  for (int kc = 0; kc < 16; ++kc) {
    ABFrag a[4];
    #pragma unroll
    for (int rb = 0; rb < 4; ++rb) {
      a[rb].h[0] = *(const v8bf*)&sH[rb * 16 + rlan][kc * 32 + kb];
      a[rb].h[1] = *(const v8bf*)&sH[rb * 16 + rlan][kc * 32 + kb + 8];
    }
    #pragma unroll
    for (int j = 0; j < 2; ++j) {
      int nt = wv * 2 + j;
      const __bf16* fp = w2b + ((size_t)(kc * 16 + nt)) * 512 + lane * 16;
      ABFrag b;
      b.h[0] = *(const v8bf*)fp;
      b.h[1] = *(const v8bf*)(fp + 8);
      #pragma unroll
      for (int rb = 0; rb < 4; ++rb)
        acc2[j][rb] = __builtin_amdgcn_wmma_f32_16x16x32_bf16(
            false, a[rb].v, false, b.v, (short)0, acc2[j][rb], false, false);
    }
  }

  // ---- bias + fused rotate_chord scatter + residual ----
  // Column tile nt (16 wide) == track nt; shift s = 2^(nt-1) (nt=0 -> 0).
  // out[b, (l-s) mod L, c] = y[b, l, c] + b2[c] + data[b, (l-s) mod L, c]
  #pragma unroll
  for (int j = 0; j < 2; ++j) {
    int nt = wv * 2 + j;
    int c  = nt * 16 + rlan;
    float bias = b2[c];
    int s = nt ? (1 << (nt - 1)) : 0;
    #pragma unroll
    for (int rb = 0; rb < 4; ++rb) {
      #pragma unroll
      for (int r = 0; r < 8; ++r) {
        int tg   = t0 + rb * 16 + (lane >> 4) * 8 + r;
        int bidx = tg >> 14;                 // L = 16384
        int lpos = tg & (L_LEN - 1);
        int lp   = (lpos - s) & (L_LEN - 1);
        size_t dst = ((((size_t)bidx << 14) | (size_t)lp) << 8) | (size_t)c;
        out[dst] = acc2[j][rb][r] + bias + data[dst];
      }
    }
  }
}

// ---------------------------------------------------------------------------
extern "C" void kernel_launch(void* const* d_in, const int* in_sizes, int n_in,
                              void* d_out, int out_size, void* d_ws, size_t ws_size,
                              hipStream_t stream) {
  const float* data = (const float*)d_in[0];
  const float* w1   = (const float*)d_in[1];
  const float* b1   = (const float*)d_in[2];
  const float* w2   = (const float*)d_in[3];
  const float* b2   = (const float*)d_in[4];
  float* out = (float*)d_out;

  // workspace: bf16 W1 fragments (256 KB) then bf16 W2 fragments (256 KB)
  __bf16* w1b = (__bf16*)d_ws;
  __bf16* w2b = w1b + (size_t)D_DIM * H_DIM;

  // 512 fragment-waves, 8 waves per 256-thread block -> 64 blocks
  prep_weights_kernel<<<64, 256, 0, stream>>>(w1, w2, w1b, w2b);

  // 65536 tokens / 64-token tiles = 1024 blocks
  chordmixer_fused_kernel<<<1024, 256, 0, stream>>>(data, b1, b2, w1b, w2b, out);
}
